// DDMHSA_87342454931955
// MI455X (gfx1250) — matrix operand order repaired and checked by reading, not compiled, not run on hardware
//
#include <hip/hip_runtime.h>

// ---------------------------------------------------------------------------
// DDMHSA for MI455X (gfx1250, wave32, WMMA bf16, async-LDS + TDM staging)
//
// Shapes: B=16, C=512, H=W=64 -> N=4096 tokens, HEADS=8, D=64, 3C=1536
// ---------------------------------------------------------------------------

typedef __attribute__((ext_vector_type(16))) __bf16 v16bf;
typedef __attribute__((ext_vector_type(8)))  __bf16 v8bf;
typedef __attribute__((ext_vector_type(8)))  float  v8f;
typedef __attribute__((ext_vector_type(4)))  float  v4f;
typedef __attribute__((ext_vector_type(4)))  unsigned int u32x4;
typedef __attribute__((ext_vector_type(8)))  int i32x8;
typedef __attribute__((ext_vector_type(4)))  int i32x4;

typedef __attribute__((address_space(1))) i32x4 gv4i;    // global v4i
typedef __attribute__((address_space(3))) i32x4 lv4i;    // LDS v4i
typedef __attribute__((address_space(3))) void  l_void;  // LDS (for offset math)

#if defined(__HIP_DEVICE_COMPILE__) && __has_builtin(__builtin_amdgcn_global_load_async_to_lds_b128)
#define HAVE_ASYNC 1
#else
#define HAVE_ASYNC 0
#endif
#if defined(__HIP_DEVICE_COMPILE__) && __has_builtin(__builtin_amdgcn_tensor_load_to_lds) && \
    __has_builtin(__builtin_amdgcn_s_wait_tensorcnt)
#define HAVE_TDM 1
#else
#define HAVE_TDM 0
#endif

#if HAVE_ASYNC
#if __has_builtin(__builtin_amdgcn_s_wait_asynccnt)
#define WAIT_ASYNC(n) __builtin_amdgcn_s_wait_asynccnt(n)
#else
#define WAIT_ASYNC(n) asm volatile("s_wait_asynccnt %0" ::"n"(n) : "memory")
#endif
#endif

#define NB   16
#define NC   512
#define NTOK 4096
#define N3C  1536
#define NH   8
#define ND   64

static __device__ __forceinline__ v16bf cat8(v8bf lo, v8bf hi) {
  return __builtin_shufflevector(lo, hi, 0,1,2,3,4,5,6,7,8,9,10,11,12,13,14,15);
}
static __device__ __forceinline__ v8f vzero8() {
  v8f z;
#pragma unroll
  for (int i = 0; i < 8; ++i) z[i] = 0.f;
  return z;
}
#define WMMA_BF16(a,b,c) \
  __builtin_amdgcn_wmma_f32_16x16x32_bf16(false, (a), false, (b), (short)0, (c), false, false)

#if HAVE_TDM
// TDM: load a 2D tile (32 K-elems x 128 rows, bf16) into LDS with row padding
// of 16B per 64B row -> LDS row stride 40 elements, matching the GEMM layout.
static __device__ __forceinline__ void tdm_load_A_tile(unsigned lds_off,
                                                       const __bf16* gsrc,
                                                       unsigned k_stride_elems) {
  unsigned long long ga = (unsigned long long)gsrc;
  u32x4 g0;
  g0[0] = 1u;                                                   // count=1, user D#
  g0[1] = lds_off;                                              // lds_addr
  g0[2] = (unsigned)ga;                                         // global_addr[31:0]
  g0[3] = (unsigned)((ga >> 32) & 0x01FFFFFFull) | (2u << 30);  // addr[56:32], type=2
  i32x8 g1;
  g1[0] = (int)((1u << 16)      // data_size = 2 bytes
              | (1u << 20)      // pad_enable
              | (3u << 22)      // pad_interval: 16 DWORDs (= one 64B row)
              | (3u << 25));    // pad_amount:   4 DWORDs (= 16B)
  g1[1] = 0;                    // tensor_dim0 = 1<<30 -> low16 = 0
  g1[2] = 0x4000;               // tensor_dim0 hi16 ; tensor_dim1 low16 = 0
  g1[3] = (32 << 16) | 0x4000;  // tensor_dim1 hi16 ; tile_dim0 = 32
  g1[4] = 128;                  // tile_dim1 = 128, tile_dim2 = 0
  g1[5] = (int)k_stride_elems;  // tensor_dim0_stride
  g1[6] = 0;
  g1[7] = 0;
  i32x4 g2, g3;
  i32x8 g4;
#pragma unroll
  for (int i = 0; i < 4; ++i) { g2[i] = 0; g3[i] = 0; }
#pragma unroll
  for (int i = 0; i < 8; ++i) g4[i] = 0;
  __builtin_amdgcn_tensor_load_to_lds(g0, g1, g2, g3, g4, 0);
}
#endif

// ---------------------------------------------------------------------------
// 1. weights -> bf16   (w_qkv: 1536x512, w_out: 512x512)
// ---------------------------------------------------------------------------
__global__ __launch_bounds__(256) void DD_convert_w(const float* __restrict__ wqkv,
                                                    const float* __restrict__ wout,
                                                    __bf16* __restrict__ wq,
                                                    __bf16* __restrict__ wo) {
  int idx = blockIdx.x * 256 + threadIdx.x;
  const int NQ = N3C * NC;
  if (idx < NQ) wq[idx] = (__bf16)wqkv[idx];
  else          wo[idx - NQ] = (__bf16)wout[idx - NQ];
}

// ---------------------------------------------------------------------------
// 2. x [b][c][p] f32 -> Xt [b][p][c] bf16 (LDS 32x32 tile transpose)
// ---------------------------------------------------------------------------
__global__ __launch_bounds__(256) void DD_transpose_x(const float* __restrict__ x,
                                                      __bf16* __restrict__ xt) {
  __shared__ float tile[32][33];
  int b = blockIdx.z, c0 = blockIdx.y * 32, p0 = blockIdx.x * 32;
  int tx = threadIdx.x & 31, ty = threadIdx.x >> 5;
#pragma unroll
  for (int r = ty; r < 32; r += 8)
    tile[r][tx] = x[((long)b * NC + c0 + r) * NTOK + p0 + tx];
  __syncthreads();
#pragma unroll
  for (int r = ty; r < 32; r += 8)
    xt[((long)b * NTOK + p0 + r) * NC + c0 + tx] = (__bf16)tile[tx][r];
}

// ---------------------------------------------------------------------------
// 3/8. Tiled bf16 WMMA GEMM:  C = A[M][K] * Bt[b][N][K]^T
//   block 256 = 8 waves, tile 128x128, k-step 32, wave tile 64x32
//   A tile staged via TDM (padded rows), B tile via async-to-LDS, dbl-buffered
//   OUT_T ? C[b][N][M] (contiguous v8f per-lane stores)  :  C[b][M][N]
// ---------------------------------------------------------------------------
template <bool OUT_T>
__global__ __launch_bounds__(256) void DD_gemm_bf16(const __bf16* __restrict__ A,
                                                    const __bf16* __restrict__ Bt,
                                                    float* __restrict__ C,
                                                    int M, int N, int K) {
  constexpr int LD = 40;                    // 80B row stride: 16B-aligned, conflict-free
  __shared__ __bf16 As[2][128 * LD];
  __shared__ __bf16 Bs[2][128 * LD];

  const int b  = blockIdx.z;
  const int m0 = blockIdx.y * 128;
  const int n0 = blockIdx.x * 128;
  const int t  = threadIdx.x;
  const int wave = t >> 5, lane = t & 31;
  const int lhalf = lane >> 4, l16 = lane & 15;
  const int wm0 = (wave >> 2) * 64;
  const int wn0 = (wave & 3) * 32;
  const __bf16* Bb = Bt + (long)b * N * K;

  const int r0  = (t * 8) >> 5;             // row covered at it=0 (it=1 adds 64)
  const int cc0 = (t * 8) & 31;             // 8-elem (16B) column chunk

  v8f acc[4][2];
#pragma unroll
  for (int i = 0; i < 4; ++i)
#pragma unroll
    for (int j = 0; j < 2; ++j) acc[i][j] = vzero8();

  auto compute_tile = [&](int buf) {
    v16bf af[4], bfv[2];
#pragma unroll
    for (int mt = 0; mt < 4; ++mt) {
      const __bf16* pa = &As[buf][(wm0 + mt * 16 + l16) * LD + lhalf * 8];
      af[mt] = cat8(*(const v8bf*)pa, *(const v8bf*)(pa + 16));
    }
#pragma unroll
    for (int nt = 0; nt < 2; ++nt) {
      const __bf16* pb = &Bs[buf][(wn0 + nt * 16 + l16) * LD + lhalf * 8];
      bfv[nt] = cat8(*(const v8bf*)pb, *(const v8bf*)(pb + 16));
    }
#pragma unroll
    for (int mt = 0; mt < 4; ++mt)
#pragma unroll
      for (int nt = 0; nt < 2; ++nt)
        acc[mt][nt] = WMMA_BF16(af[mt], bfv[nt], acc[mt][nt]);
  };

#if HAVE_ASYNC
  auto stage = [&](int buf, int k0) {
#if HAVE_TDM
    if (wave == 0)
      tdm_load_A_tile((unsigned)(unsigned long long)(l_void*)&As[buf][0],
                      A + (long)m0 * K + k0, (unsigned)K);
#else
#pragma unroll
    for (int it = 0; it < 2; ++it) {
      int row = r0 + it * 64;
      __builtin_amdgcn_global_load_async_to_lds_b128(
          (gv4i*)(unsigned long long)(A + (long)(m0 + row) * K + k0 + cc0),
          (lv4i*)(unsigned)(unsigned long long)&As[buf][row * LD + cc0], 0, 0);
    }
#endif
#pragma unroll
    for (int it = 0; it < 2; ++it) {
      int row = r0 + it * 64;
      __builtin_amdgcn_global_load_async_to_lds_b128(
          (gv4i*)(unsigned long long)(Bb + (long)(n0 + row) * K + k0 + cc0),
          (lv4i*)(unsigned)(unsigned long long)&Bs[buf][row * LD + cc0], 0, 0);
    }
  };

  const int nk = K >> 5;
  stage(0, 0);
  for (int kt = 0; kt < nk - 1; ++kt) {
    stage((kt + 1) & 1, (kt + 1) << 5);     // prefetch next tile into other buffer
#if HAVE_TDM
    if (wave == 0) __builtin_amdgcn_s_wait_tensorcnt(1);
    WAIT_ASYNC(2);
#else
    WAIT_ASYNC(4);
#endif
    __syncthreads();
    compute_tile(kt & 1);
    __syncthreads();
  }
#if HAVE_TDM
  if (wave == 0) __builtin_amdgcn_s_wait_tensorcnt(0);
#endif
  WAIT_ASYNC(0);
  __syncthreads();
  compute_tile((nk - 1) & 1);
#else
  // fallback: synchronous single-buffer staging
  for (int k0 = 0; k0 < K; k0 += 32) {
#pragma unroll
    for (int it = 0; it < 2; ++it) {
      int row = r0 + it * 64;
      *(v8bf*)(&As[0][row * LD + cc0]) = *(const v8bf*)(A  + (long)(m0 + row) * K + k0 + cc0);
      *(v8bf*)(&Bs[0][row * LD + cc0]) = *(const v8bf*)(Bb + (long)(n0 + row) * K + k0 + cc0);
    }
    __syncthreads();
    compute_tile(0);
    __syncthreads();
  }
#endif

  // epilogue: D layout = lane holds col n=l16, rows m = lhalf*8 + j (8 consecutive)
#pragma unroll
  for (int mt = 0; mt < 4; ++mt)
#pragma unroll
    for (int nt = 0; nt < 2; ++nt) {
      int mbase = m0 + wm0 + mt * 16 + lhalf * 8;
      int n     = n0 + wn0 + nt * 16 + l16;
      if (OUT_T) {
        *(v8f*)(C + ((long)b * N + n) * M + mbase) = acc[mt][nt];
      } else {
        float* cp = C + (long)b * M * N;
#pragma unroll
        for (int j = 0; j < 8; ++j)
          cp[(long)(mbase + j) * N + n] = acc[mt][nt][j];
      }
    }
}

// ---------------------------------------------------------------------------
// 4. depthwise 3x3, dilation 2, pad 2, NHWC [b][p][1536]
// ---------------------------------------------------------------------------
__global__ __launch_bounds__(256) void DD_dwconv(const float* __restrict__ in,
                                                 const float* __restrict__ wdw,
                                                 float* __restrict__ out) {
  int o  = blockIdx.y * 256 + threadIdx.x;
  int bp = blockIdx.x;
  int b  = bp >> 12, p = bp & 4095;
  int y  = p >> 6,  x = p & 63;
  if (y + 1 < 64)  // prefetch next spatial row (-> global_prefetch_b8)
    __builtin_prefetch(&in[((long)b * NTOK + (y + 1) * 64 + x) * N3C + o], 0, 1);
  const float* wr = wdw + (o & 511) * 9;
  float s = 0.f;
#pragma unroll
  for (int ky = 0; ky < 3; ++ky) {
    int yy = y + (ky - 1) * 2;
    if (yy < 0 || yy >= 64) continue;
#pragma unroll
    for (int kx = 0; kx < 3; ++kx) {
      int xx = x + (kx - 1) * 2;
      if (xx < 0 || xx >= 64) continue;
      s += in[((long)b * NTOK + yy * 64 + xx) * N3C + o] * wr[ky * 3 + kx];
    }
  }
  out[((long)b * NTOK + p) * N3C + o] = s;
}

// ---------------------------------------------------------------------------
// 5. q/k: L2 norm over d (contiguous 64 f32), fold 1/norm, transpose ->
//    dst[bh][d][n] bf16.  block = (b,qk) x h x n-tile(32)
// ---------------------------------------------------------------------------
__global__ __launch_bounds__(256) void DD_normalize_qk(const float* __restrict__ qkvc,
                                                       __bf16* __restrict__ qb,
                                                       __bf16* __restrict__ kb) {
  __shared__ float tile[32][68];
  __shared__ float sc[32];
  int z = blockIdx.z, b = z >> 1, qk = z & 1;
  int h = blockIdx.y, n0 = blockIdx.x * 32;
  int t = threadIdx.x;

  int nr = t >> 3, d0 = (t & 7) * 8;
  const float* src = qkvc + ((long)b * NTOK + n0 + nr) * N3C + qk * NC + h * ND + d0;
  *(v4f*)(&tile[nr][d0])     = *(const v4f*)src;
  *(v4f*)(&tile[nr][d0 + 4]) = *(const v4f*)(src + 4);
  __syncthreads();

  if (t < 32) {
    float s = 0.f;
#pragma unroll
    for (int d = 0; d < 64; ++d) { float v = tile[t][d]; s += v * v; }
    sc[t] = 1.f / fmaxf(__builtin_sqrtf(s), 1e-12f);
  }
  __syncthreads();

  __bf16* dst = qk ? kb : qb;
  long base = (long)(b * NH + h) * ND * NTOK;
  int n = t & 31;
#pragma unroll
  for (int u = 0; u < 8; ++u) {
    int d = (t >> 5) * 8 + u;
    dst[base + (long)d * NTOK + n0 + n] = (__bf16)(tile[n][d] * sc[n]);
  }
}

// ---------------------------------------------------------------------------
// 6. v: [b][n][1024+h*64+e] f32 -> vb[bh][n][e] bf16
// ---------------------------------------------------------------------------
__global__ __launch_bounds__(256) void DD_convert_v(const float* __restrict__ qkvc,
                                                    __bf16* __restrict__ vbt) {
  long idx = ((long)blockIdx.x * 256 + threadIdx.x) * 8;
  int  e = (int)(idx & 63);
  long r = idx >> 6;
  int  n = (int)(r & 4095);
  int  bh = (int)(r >> 12), b = bh >> 3, h = bh & 7;
  const float* src = qkvc + ((long)b * NTOK + n) * N3C + 2 * NC + h * ND + e;
  v4f a = *(const v4f*)src, c = *(const v4f*)(src + 4);
  v8bf o;
#pragma unroll
  for (int i = 0; i < 4; ++i) { o[i] = (__bf16)a[i]; o[4 + i] = (__bf16)c[i]; }
  *(v8bf*)(vbt + idx) = o;
}

// ---------------------------------------------------------------------------
// 7. attention: one block per (b,h).  8 waves.
// ---------------------------------------------------------------------------
__global__ __launch_bounds__(256) void DD_attention(const __bf16* __restrict__ qb,
                                                    const __bf16* __restrict__ kb,
                                                    const __bf16* __restrict__ vbt,
                                                    __bf16* __restrict__ obt,
                                                    const float* __restrict__ temp_p) {
  __shared__ float  attn_s[64 * 65];
  __shared__ __bf16 attnb[64 * 72];

  const int bh = blockIdx.x, b = bh >> 3, h = bh & 7;
  const int t = threadIdx.x, wave = t >> 5, lane = t & 31;
  const int lhalf = lane >> 4, l16 = lane & 15;
  const float tinv = 1.f / fmaxf(temp_p[0], 1e-5f);

  { // phase 1: attn = q * k^T (K = 4096)
    const int mt  = wave >> 1;
    const int nt0 = (wave & 1) * 2;
    v8f acc[2] = {vzero8(), vzero8()};
    const __bf16* qrow = qb + ((long)bh * ND + mt * 16 + l16) * NTOK;
    for (int k0 = 0; k0 < NTOK; k0 += 32) {
      v16bf af = cat8(*(const v8bf*)(qrow + k0 + lhalf * 8),
                      *(const v8bf*)(qrow + k0 + 16 + lhalf * 8));
#pragma unroll
      for (int nt = 0; nt < 2; ++nt) {
        const __bf16* krow = kb + ((long)bh * ND + (nt0 + nt) * 16 + l16) * NTOK
                             + k0 + lhalf * 8;
        v16bf bf = cat8(*(const v8bf*)krow, *(const v8bf*)(krow + 16));
        acc[nt] = WMMA_BF16(af, bf, acc[nt]);
      }
    }
#pragma unroll
    for (int nt = 0; nt < 2; ++nt) {
      int e = (nt0 + nt) * 16 + l16;
#pragma unroll
      for (int j = 0; j < 8; ++j)
        attn_s[(mt * 16 + lhalf * 8 + j) * 65 + e] = acc[nt][j] * tinv;
    }
  }
  __syncthreads();

  if (t < 64) { // softmax over e
    float m = -3.0e38f;
#pragma unroll
    for (int e = 0; e < 64; ++e) m = fmaxf(m, attn_s[t * 65 + e]);
    float s = 0.f;
#pragma unroll
    for (int e = 0; e < 64; ++e) {
      float ex = __expf(attn_s[t * 65 + e] - m);
      attn_s[t * 65 + e] = ex;
      s += ex;
    }
    float inv = 1.f / s;
#pragma unroll
    for (int e = 0; e < 64; ++e)
      attnb[t * 72 + e] = (__bf16)(attn_s[t * 65 + e] * inv);
  }
  __syncthreads();

  // phase 2: out = attnb(64x64) x V, n-tiles over waves
  v16bf af2[4][2];
#pragma unroll
  for (int mt = 0; mt < 4; ++mt)
#pragma unroll
    for (int kk = 0; kk < 2; ++kk) {
      const __bf16* pa = &attnb[(mt * 16 + l16) * 72 + kk * 32 + lhalf * 8];
      af2[mt][kk] = cat8(*(const v8bf*)pa, *(const v8bf*)(pa + 16));
    }

  for (int nt = wave; nt < 256; nt += 8) {
    const int n0 = nt * 16;
    v8f oc[4];
#pragma unroll
    for (int mt = 0; mt < 4; ++mt) oc[mt] = vzero8();
#pragma unroll
    for (int kk = 0; kk < 2; ++kk) {
      const __bf16* pv = vbt + ((long)bh * NTOK + n0 + l16) * ND + kk * 32 + lhalf * 8;
      v16bf bf = cat8(*(const v8bf*)pv, *(const v8bf*)(pv + 16));
#pragma unroll
      for (int mt = 0; mt < 4; ++mt)
        oc[mt] = WMMA_BF16(af2[mt][kk], bf, oc[mt]);
    }
#pragma unroll
    for (int mt = 0; mt < 4; ++mt) {
      int c0 = h * ND + mt * 16 + lhalf * 8;
      int n  = n0 + l16;
      v8bf o;
#pragma unroll
      for (int j = 0; j < 8; ++j) o[j] = (__bf16)oc[mt][j];
      *(v8bf*)(obt + ((long)b * NTOK + n) * NC + c0) = o;
    }
  }
}

// ---------------------------------------------------------------------------
extern "C" void kernel_launch(void* const* d_in, const int* in_sizes, int n_in,
                              void* d_out, int out_size, void* d_ws, size_t ws_size,
                              hipStream_t stream) {
  (void)in_sizes; (void)n_in; (void)out_size; (void)ws_size;
  const float* x      = (const float*)d_in[0];
  const float* w_qkv  = (const float*)d_in[1];
  const float* w_dw   = (const float*)d_in[2];
  const float* w_out  = (const float*)d_in[3];
  const float* temp_p = (const float*)d_in[4];
  float* out = (float*)d_out;

  char* ws = (char*)d_ws;
  const size_t szXt   = (size_t)NB * NTOK * NC * 2;
  const size_t szWq   = (size_t)N3C * NC * 2;
  const size_t szWo   = (size_t)NC * NC * 2;
  const size_t szQKV  = (size_t)NB * NTOK * N3C * 4;
  const size_t szHead = (size_t)NB * NH * ND * NTOK * 2;

  __bf16* Xt   = (__bf16*)(ws);
  __bf16* Wqb  = (__bf16*)(ws + szXt);
  __bf16* Wob  = (__bf16*)(ws + szXt + szWq);
  float*  QKVt = (float*) (ws + szXt + szWq + szWo);
  float*  QKVc = (float*) (ws + szXt + szWq + szWo + szQKV);
  __bf16* Qb   = (__bf16*)(ws + szXt + szWq + szWo + 2 * szQKV);
  __bf16* Kb   = (__bf16*)(ws + szXt + szWq + szWo + 2 * szQKV + szHead);
  __bf16* Vb   = (__bf16*)(ws + szXt + szWq + szWo + 2 * szQKV + 2 * szHead);
  __bf16* Ob   = (__bf16*)(ws + szXt + szWq + szWo + 2 * szQKV + 3 * szHead);

  DD_convert_w<<<4096, 256, 0, stream>>>(w_qkv, w_out, Wqb, Wob);
  DD_transpose_x<<<dim3(128, 16, NB), 256, 0, stream>>>(x, Xt);
  DD_gemm_bf16<true><<<dim3(32, 12, NB), 256, 0, stream>>>(Wqb, Xt, QKVt, N3C, NTOK, NC);
  DD_dwconv<<<dim3(NB * NTOK, 6), 256, 0, stream>>>(QKVt, w_dw, QKVc);
  DD_normalize_qk<<<dim3(128, NH, NB * 2), 256, 0, stream>>>(QKVc, Qb, Kb);
  DD_convert_v<<<16384, 256, 0, stream>>>(QKVc, Vb);
  DD_attention<<<NB * NH, 256, 0, stream>>>(Qb, Kb, Vb, Ob, temp_p);
  DD_gemm_bf16<false><<<dim3(32, 4, NB), 256, 0, stream>>>(Wob, Ob, out, NC, NTOK, NC);
}
